// NeuralMemoryModule_1254130450962
// MI455X (gfx1250) — matrix operand order repaired and checked by reading, not compiled, not run on hardware
//
#include <hip/hip_runtime.h>
#include <hip/hip_bf16.h>

// ---------------------------------------------------------------------------
// NeuralMemoryModule for MI455X (gfx1250): bf16 WMMA pipeline.
// B=16, T=8192, D=256.  M_prev = I, S_prev = 0 =>
//   k = x Wk^T, v = x Wv^T, diff = k - v
//   G = k^T diff * 2/(B*T*D) per batch, S = -0.1*G, M = (1-alpha)I + S
//   out = LN(x M)
// ---------------------------------------------------------------------------

#define B_DIM 16
#define T_DIM 8192
#define D_DIM 256

typedef __bf16 bf16_t;
typedef __bf16 v16bf __attribute__((ext_vector_type(16)));
typedef __bf16 v8bf  __attribute__((ext_vector_type(8)));
typedef float  v8f   __attribute__((ext_vector_type(8)));
typedef int    v4i   __attribute__((ext_vector_type(4)));

union A16 { v16bf v; bf16_t e[16]; };
union F8  { v8f   v; float  e[8];  };

// ---- CDNA5 async global->LDS copy (ASYNCcnt path), guarded by __has_builtin.
#if defined(__has_builtin)
# if __has_builtin(__builtin_amdgcn_global_load_async_to_lds_b128) && \
     __has_builtin(__builtin_amdgcn_s_wait_asynccnt)
#  define USE_ASYNC_LDS 1
# endif
#endif

typedef __attribute__((address_space(1))) v4i g_v4i;
typedef __attribute__((address_space(3))) v4i l_v4i;

__device__ __forceinline__ void cp16_g2l(const bf16_t* g, bf16_t* l) {
#ifdef USE_ASYNC_LDS
    __builtin_amdgcn_global_load_async_to_lds_b128(
        (g_v4i*)(uintptr_t)g,
        (l_v4i*)(uint32_t)(uintptr_t)l,
        /*offset=*/0, /*cpol=*/0);
#else
    *(v8bf*)l = *(const v8bf*)g;
#endif
}

__device__ __forceinline__ void lds_wait_barrier() {
#ifdef USE_ASYNC_LDS
    __builtin_amdgcn_s_wait_asynccnt(0);
#endif
    __syncthreads();
}

__device__ __forceinline__ v8f wmma_bf16(v16bf a, v16bf b, v8f c) {
    // v_wmma_f32_16x16x32_bf16: D = A(16x32) x B(32x16) + C(16x16 f32)
    return __builtin_amdgcn_wmma_f32_16x16x32_bf16(
        /*neg_a=*/false, a, /*neg_b=*/false, b,
        /*c_mod=*/(short)0, c, /*reuse_a=*/false, /*reuse_b=*/false);
}

// A-matrix (16x32 bf16, M x K) per-lane fragment from a row-major f32 source.
// Lane L: M = L&15, h = L>>4.  a[0..7] = row[K0 + h*8 + j], a[8..15] = row[K0+16+h*8+j].
__device__ __forceinline__ v16bf load_a_f32(const float* __restrict__ row,
                                            int k0, int lane) {
    const int h = (lane >> 4) << 3;       // 0 or 8
    const float4* p0 = (const float4*)(row + k0 + h);
    const float4* p1 = (const float4*)(row + k0 + 16 + h);
    float4 x0 = p0[0], x1 = p0[1];
    float4 y0 = p1[0], y1 = p1[1];
    A16 u;
    u.e[0]  = (bf16_t)x0.x; u.e[1]  = (bf16_t)x0.y;
    u.e[2]  = (bf16_t)x0.z; u.e[3]  = (bf16_t)x0.w;
    u.e[4]  = (bf16_t)x1.x; u.e[5]  = (bf16_t)x1.y;
    u.e[6]  = (bf16_t)x1.z; u.e[7]  = (bf16_t)x1.w;
    u.e[8]  = (bf16_t)y0.x; u.e[9]  = (bf16_t)y0.y;
    u.e[10] = (bf16_t)y0.z; u.e[11] = (bf16_t)y0.w;
    u.e[12] = (bf16_t)y1.x; u.e[13] = (bf16_t)y1.y;
    u.e[14] = (bf16_t)y1.z; u.e[15] = (bf16_t)y1.w;
    return u.v;
}

// ---------------------------------------------------------------------------
// Kernel 1: convert Wk, Wv to bf16 row-major [out][in] (B-operand layout).
__global__ void __launch_bounds__(256)
k_convert_w(const float* __restrict__ Wk, const float* __restrict__ Wv,
            bf16_t* __restrict__ WkT, bf16_t* __restrict__ WvT) {
    int idx = blockIdx.x * 256 + threadIdx.x;     // 65536 threads
    int o = idx >> 8, i = idx & 255;
    WkT[o * D_DIM + i] = (bf16_t)Wk[o * D_DIM + i];
    WvT[o * D_DIM + i] = (bf16_t)Wv[o * D_DIM + i];
}

// ---------------------------------------------------------------------------
// Kernel 2: column sums of x over T -> colsum[b][d] (for x.mean(axis=1)).
__global__ void __launch_bounds__(256)
k_colsum(const float* __restrict__ x, float* __restrict__ colsum) {
    __shared__ float red[256];
    int b = blockIdx.x >> 8;
    int d = blockIdx.x & 255;
    float s = 0.f;
    for (int t = threadIdx.x; t < T_DIM; t += 256)
        s += x[((size_t)b * T_DIM + t) * D_DIM + d];
    red[threadIdx.x] = s;
    __syncthreads();
#pragma unroll
    for (int off = 128; off > 0; off >>= 1) {
        if (threadIdx.x < off) red[threadIdx.x] += red[threadIdx.x + off];
        __syncthreads();
    }
    if (threadIdx.x == 0) colsum[b * D_DIM + d] = red[0];
}

// Kernel 3: alpha[b] = sigmoid(mean_x[b] . gate_w + gate_b)
__global__ void __launch_bounds__(256)
k_alpha(const float* __restrict__ colsum, const float* __restrict__ gw,
        const float* __restrict__ gb, float* __restrict__ alpha) {
    __shared__ float red[256];
    int b = blockIdx.x, d = threadIdx.x;
    red[d] = (colsum[b * D_DIM + d] * (1.0f / T_DIM)) * gw[d];
    __syncthreads();
#pragma unroll
    for (int off = 128; off > 0; off >>= 1) {
        if (d < off) red[d] += red[d + off];
        __syncthreads();
    }
    if (d == 0) {
        float logit = red[0] + gb[0];
        alpha[b] = 1.0f / (1.0f + expf(-logit));
    }
}

// ---------------------------------------------------------------------------
// Kernel 4: k = x Wk^T, v = x Wv^T; emit kT (bf16 [B,D,T]) and diff (bf16 [B,T,D]).
// Block = 16 T-rows x 256 outs; 8 waves, each wave owns 2 N-tiles for both k,v.
__global__ void __launch_bounds__(256)
k_gemm_kv(const float* __restrict__ x, const bf16_t* __restrict__ WkT,
          const bf16_t* __restrict__ WvT, bf16_t* __restrict__ kT,
          bf16_t* __restrict__ diff) {
    const int blk = blockIdx.x;                 // 8192 blocks
    const int b  = blk >> 9;
    const int t0 = (blk & 511) << 4;
    const int tid  = threadIdx.x;
    const int wave = tid >> 5, lane = tid & 31;
    const int n0 = wave << 5;                   // 2 N-tiles: n0, n0+16
    const int m = lane & 15, h = lane >> 4, col = lane & 15;

    const float* arow = x + ((size_t)b * T_DIM + t0 + m) * D_DIM;

    v8f ka0 = {}, ka1 = {}, va0 = {}, va1 = {};
#pragma unroll
    for (int kk = 0; kk < D_DIM; kk += 32) {
        v16bf a = load_a_f32(arow, kk, lane);
        const bf16_t* bk = WkT + (size_t)(n0 + col) * D_DIM + kk + (h << 3);
        const bf16_t* bv = WvT + (size_t)(n0 + col) * D_DIM + kk + (h << 3);
        v16bf bk0 = __builtin_shufflevector(*(const v8bf*)bk, *(const v8bf*)(bk + 16), 0,1,2,3,4,5,6,7,8,9,10,11,12,13,14,15);
        v16bf bk1 = __builtin_shufflevector(*(const v8bf*)(bk + 16 * D_DIM), *(const v8bf*)(bk + 16 * D_DIM + 16), 0,1,2,3,4,5,6,7,8,9,10,11,12,13,14,15);
        v16bf bv0 = __builtin_shufflevector(*(const v8bf*)bv, *(const v8bf*)(bv + 16), 0,1,2,3,4,5,6,7,8,9,10,11,12,13,14,15);
        v16bf bv1 = __builtin_shufflevector(*(const v8bf*)(bv + 16 * D_DIM), *(const v8bf*)(bv + 16 * D_DIM + 16), 0,1,2,3,4,5,6,7,8,9,10,11,12,13,14,15);
        ka0 = wmma_bf16(a, bk0, ka0);
        ka1 = wmma_bf16(a, bk1, ka1);
        va0 = wmma_bf16(a, bv0, va0);
        va1 = wmma_bf16(a, bv1, va1);
    }

    F8 K0, K1, V0, V1;
    K0.v = ka0; K1.v = ka1; V0.v = va0; V1.v = va1;
    bf16_t* kTb = kT + (size_t)b * D_DIM * T_DIM;
    bf16_t* dfb = diff + (size_t)b * T_DIM * D_DIM;
#pragma unroll
    for (int r = 0; r < 8; ++r) {
        const int t  = t0 + r + (h << 3);
        const int c0 = n0 + col, c1 = n0 + 16 + col;
        float k0v = K0.e[r], k1v = K1.e[r];
        kTb[(size_t)c0 * T_DIM + t] = (bf16_t)k0v;
        kTb[(size_t)c1 * T_DIM + t] = (bf16_t)k1v;
        dfb[(size_t)t * D_DIM + c0] = (bf16_t)(k0v - V0.e[r]);
        dfb[(size_t)t * D_DIM + c1] = (bf16_t)(k1v - V1.e[r]);
    }
}

// ---------------------------------------------------------------------------
// Kernel 5: G = kT . diff over T; S = -0.1*scale*G; M = (1-alpha)I + S.
// LDS-tiled, async double-buffered: workgroup = 64x64 output, 4 waves, each
// wave owns one 16-row band x four 16-col tiles.  K = 8192 in steps of 32.
__global__ void __launch_bounds__(128)
k_grad(const bf16_t* __restrict__ kT, const bf16_t* __restrict__ diff,
       const float* __restrict__ alpha, float* __restrict__ Mt_out,
       float* __restrict__ St_out, bf16_t* __restrict__ Mt_bf) {
    // Row strides padded but kept 16B-aligned: A row = 80B (20 dwords),
    // B row = 144B (36 dwords).
    __shared__ __align__(16) bf16_t As[2][64][40];
    __shared__ __align__(16) bf16_t Bs[2][32][72];

    const int b  = blockIdx.y;
    const int m0 = (blockIdx.x & 3) << 6;
    const int n0 = (blockIdx.x >> 2) << 6;
    const int tid  = threadIdx.x;
    const int wave = tid >> 5, lane = tid & 31;
    const int h = lane >> 4, mrow = lane & 15, col = lane & 15;

    const bf16_t* Ab = kT + (size_t)b * D_DIM * T_DIM;
    const bf16_t* Bb = diff + (size_t)b * T_DIM * D_DIM;

    // Cooperative load mapping (128 threads):
    //  A slab 64x32: thread -> row tid>>1, 16-elem chunk (tid&1)*16  (2 x b128)
    //  B slab 32x64: thread -> row tid>>2, 16-elem chunk (tid&3)*16  (2 x b128)
    const int ar = tid >> 1, ac = (tid & 1) << 4;
    const int br = tid >> 2, bc = (tid & 3) << 4;

    auto issue = [&](int buf, int k0) {
        const bf16_t* ga = Ab + (size_t)(m0 + ar) * T_DIM + k0 + ac;
        cp16_g2l(ga,     &As[buf][ar][ac]);
        cp16_g2l(ga + 8, &As[buf][ar][ac + 8]);
        const bf16_t* gb = Bb + (size_t)(k0 + br) * D_DIM + n0 + bc;
        cp16_g2l(gb,     &Bs[buf][br][bc]);
        cp16_g2l(gb + 8, &Bs[buf][br][bc + 8]);
    };

    v8f acc[4] = {{}, {}, {}, {}};

    issue(0, 0);
    lds_wait_barrier();

    const int NK = T_DIM / 32;
    for (int kk = 0; kk < NK; ++kk) {
        const int buf = kk & 1;
        if (kk + 1 < NK) issue(buf ^ 1, (kk + 1) << 5);

        // A fragment (16x32) for this wave's M-band, served from LDS.
        const bf16_t* arow = &As[buf][(wave << 4) + mrow][0];
        v8bf lo = *(const v8bf*)(arow + (h << 3));
        v8bf hi = *(const v8bf*)(arow + 16 + (h << 3));
        A16 ua;
#pragma unroll
        for (int j = 0; j < 8; ++j) { ua.e[j] = lo[j]; ua.e[8 + j] = hi[j]; }

        // B fragments: lane L holds column L&15 of each 32x16 tile, K in table order.
#pragma unroll
        for (int nt = 0; nt < 4; ++nt) {
            A16 ub;
#pragma unroll
            for (int j = 0; j < 8; ++j) {
                ub.e[j]     = Bs[buf][(h << 3) + j][(nt << 4) + col];
                ub.e[8 + j] = Bs[buf][16 + (h << 3) + j][(nt << 4) + col];
            }
            acc[nt] = wmma_bf16(ua.v, ub.v, acc[nt]);
        }

        lds_wait_barrier();   // loads for buf^1 done + everyone done reading buf
    }

    const float scale = -(1.0f - 0.9f) * 2.0f /
                        ((float)B_DIM * (float)T_DIM * (float)D_DIM);
    const float oma = 1.0f - alpha[b];
#pragma unroll
    for (int nt = 0; nt < 4; ++nt) {
        F8 G; G.v = acc[nt];
#pragma unroll
        for (int r = 0; r < 8; ++r) {
            const int row = m0 + (wave << 4) + r + (h << 3);
            const int cc  = n0 + (nt << 4) + col;
            float s  = G.e[r] * scale;
            float mt = (row == cc ? oma : 0.0f) + s;
            size_t idx = ((size_t)b * D_DIM + row) * D_DIM + cc;
            St_out[idx] = s;
            Mt_out[idx] = mt;
            Mt_bf[((size_t)b * D_DIM + cc) * D_DIM + row] = (bf16_t)mt;
        }
    }
}

// ---------------------------------------------------------------------------
// Kernel 6: out = LayerNorm(x . M_t). Block = 16 T-rows x full D=256 (8 waves,
// 2 N-tiles each), fused row-wise LN via LDS.
__global__ void __launch_bounds__(256)
k_out_ln(const float* __restrict__ x, const bf16_t* __restrict__ Mt,
         const float* __restrict__ ln_g, const float* __restrict__ ln_b,
         float* __restrict__ out) {
    __shared__ float sm[16 * 264];           // padded stride to dodge banks
    __shared__ float mu_s[16], rs_s[16];

    const int blk = blockIdx.x;              // 8192 blocks
    const int b  = blk >> 9;
    const int t0 = (blk & 511) << 4;
    const int tid  = threadIdx.x;
    const int wave = tid >> 5, lane = tid & 31;
    const int n0 = wave << 5;
    const int m = lane & 15, h = lane >> 4, col = lane & 15;

    const float*  arow = x + ((size_t)b * T_DIM + t0 + m) * D_DIM;
    const bf16_t* bbuf = Mt + (size_t)b * D_DIM * D_DIM;

    v8f a0 = {}, a1 = {};
#pragma unroll
    for (int kk = 0; kk < D_DIM; kk += 32) {
        v16bf a = load_a_f32(arow, kk, lane);
        const bf16_t* brow = bbuf + (size_t)(n0 + col) * D_DIM + kk + (h << 3);
        v16bf b0 = __builtin_shufflevector(*(const v8bf*)brow, *(const v8bf*)(brow + 16), 0,1,2,3,4,5,6,7,8,9,10,11,12,13,14,15);
        v16bf b1 = __builtin_shufflevector(*(const v8bf*)(brow + 16 * D_DIM), *(const v8bf*)(brow + 16 * D_DIM + 16), 0,1,2,3,4,5,6,7,8,9,10,11,12,13,14,15);
        a0 = wmma_bf16(a, b0, a0);
        a1 = wmma_bf16(a, b1, a1);
    }

    F8 H0, H1; H0.v = a0; H1.v = a1;
#pragma unroll
    for (int r = 0; r < 8; ++r) {
        const int row = r + (h << 3);
        sm[row * 264 + n0 + col]      = H0.e[r];
        sm[row * 264 + n0 + 16 + col] = H1.e[r];
    }
    __syncthreads();

    if (tid < 16) {
        float s = 0.f, ss = 0.f;
        for (int e = 0; e < D_DIM; ++e) {
            float v = sm[tid * 264 + e];
            s += v; ss += v * v;
        }
        float mu  = s * (1.0f / D_DIM);
        float var = ss * (1.0f / D_DIM) - mu * mu;
        mu_s[tid] = mu;
        rs_s[tid] = rsqrtf(var + 1e-5f);
    }
    __syncthreads();

    const int row = tid >> 4, cb = (tid & 15) << 4;
    const float mu = mu_s[row], rs = rs_s[row];
    float* orow = out + ((size_t)b * T_DIM + t0 + row) * D_DIM;
#pragma unroll
    for (int j = 0; j < 16; ++j) {
        const int e = cb + j;
        float hv = sm[row * 264 + e];
        orow[e] = (hv - mu) * rs * ln_g[e] + ln_b[e];
    }
}

// ---------------------------------------------------------------------------
extern "C" void kernel_launch(void* const* d_in, const int* in_sizes, int n_in,
                              void* d_out, int out_size, void* d_ws, size_t ws_size,
                              hipStream_t stream) {
    (void)in_sizes; (void)n_in; (void)out_size; (void)ws_size;

    const float* x      = (const float*)d_in[0];
    const float* Wk     = (const float*)d_in[1];
    const float* Wv     = (const float*)d_in[2];
    const float* gate_w = (const float*)d_in[3];
    const float* gate_b = (const float*)d_in[4];
    const float* ln_g   = (const float*)d_in[5];
    const float* ln_b   = (const float*)d_in[6];
    // d_in[7] = current_step (always 0 path)

    float* out_f = (float*)d_out;
    float* out_o = out_f;                                        // [B,T,D]
    float* Mt_o  = out_f + (size_t)B_DIM * T_DIM * D_DIM;        // [B,D,D]
    float* St_o  = Mt_o  + (size_t)B_DIM * D_DIM * D_DIM;        // [B,D,D]

    char* ws = (char*)d_ws;
    size_t off = 0;
    bf16_t* kT_ws   = (bf16_t*)(ws + off); off += (size_t)B_DIM * D_DIM * T_DIM * 2;  // 64 MB
    bf16_t* diff_ws = (bf16_t*)(ws + off); off += (size_t)B_DIM * T_DIM * D_DIM * 2;  // 64 MB
    bf16_t* Mt_ws   = (bf16_t*)(ws + off); off += (size_t)B_DIM * D_DIM * D_DIM * 2;  //  2 MB
    bf16_t* WkT_ws  = (bf16_t*)(ws + off); off += (size_t)D_DIM * D_DIM * 2;
    bf16_t* WvT_ws  = (bf16_t*)(ws + off); off += (size_t)D_DIM * D_DIM * 2;
    float*  colsum  = (float*)(ws + off);  off += (size_t)B_DIM * D_DIM * 4;
    float*  alpha   = (float*)(ws + off);  off += (size_t)B_DIM * 4;

    k_convert_w<<<256, 256, 0, stream>>>(Wk, Wv, WkT_ws, WvT_ws);
    k_colsum<<<B_DIM * D_DIM, 256, 0, stream>>>(x, colsum);
    k_alpha<<<B_DIM, 256, 0, stream>>>(colsum, gate_w, gate_b, alpha);
    k_gemm_kv<<<(B_DIM * T_DIM) / 16, 256, 0, stream>>>(x, WkT_ws, WvT_ws,
                                                        kT_ws, diff_ws);
    dim3 gGrad(16, B_DIM);
    k_grad<<<gGrad, 128, 0, stream>>>(kT_ws, diff_ws, alpha, Mt_o, St_o, Mt_ws);
    k_out_ln<<<(B_DIM * T_DIM) / 16, 256, 0, stream>>>(x, Mt_ws, ln_g, ln_b, out_o);
}
